// GroupedQueryAttention_11785390260959
// MI455X (gfx1250) — matrix operand order repaired and checked by reading, not compiled, not run on hardware
//
#include <hip/hip_runtime.h>
#include <hip/hip_bf16.h>
#include <math.h>

typedef __bf16 bf16;
typedef __attribute__((ext_vector_type(16))) __bf16 v16bf;
typedef __attribute__((ext_vector_type(8)))  float   v8f;

#define L_SEQ   2048
#define D_MODEL 2048
#define QKV_DIM 3072
#define N_HQ    16
#define N_HK    4
#define HD      128
#define WINDOW  1024

#if defined(__has_builtin)
#if __has_builtin(__builtin_amdgcn_global_load_async_to_lds_b128)
#define HAVE_ASYNC_LDS 1
#endif
#endif

union BF16Frag { v16bf v; uint4 q[2]; };

__device__ __forceinline__ v16bf ld_frag_contig(const bf16* p) {
  BF16Frag u;
  u.q[0] = ((const uint4*)p)[0];
  u.q[1] = ((const uint4*)p)[1];
  return u.v;
}
__device__ __forceinline__ v16bf ld_frag_2x8(const bf16* p0, const bf16* p1) {
  BF16Frag u;
  u.q[0] = *(const uint4*)p0;
  u.q[1] = *(const uint4*)p1;
  return u.v;
}
__device__ __forceinline__ v8f v8f_zero() {
  v8f z;
#pragma unroll
  for (int e = 0; e < 8; ++e) z[e] = 0.0f;
  return z;
}
__device__ __forceinline__ v8f wmma_bf16(v16bf a, v16bf b, v8f c) {
  return __builtin_amdgcn_wmma_f32_16x16x32_bf16(false, a, false, b, (short)0, c,
                                                 false, false);
}

// 16-byte memory->LDS copy: async (no VGPR round trip, ASYNCcnt) when available
__device__ __forceinline__ void cp_lds_b128(const bf16* g, bf16* l) {
#ifdef HAVE_ASYNC_LDS
  typedef int v4i __attribute__((vector_size(16)));
  typedef __attribute__((address_space(1))) v4i* gvec_p;
  typedef __attribute__((address_space(3))) v4i* lvec_p;
  gvec_p gp = (gvec_p)(__attribute__((address_space(1))) void*)(void*)g;
  lvec_p lp = (lvec_p)(__attribute__((address_space(3))) void*)l;
  __builtin_amdgcn_global_load_async_to_lds_b128(gp, lp, 0, 0);
#else
  *(uint4*)l = *(const uint4*)g;
#endif
}
__device__ __forceinline__ void cp_lds_wait() {
#ifdef HAVE_ASYNC_LDS
#if __has_builtin(__builtin_amdgcn_s_wait_asynccnt)
  __builtin_amdgcn_s_wait_asynccnt(0);
#else
  asm volatile("s_wait_asynccnt 0" ::: "memory");
#endif
#endif
}

// ---------------------------------------------------------------------------
// f32 -> bf16 conversion
// ---------------------------------------------------------------------------
__global__ void cvt_f32_bf16_kernel(const float* __restrict__ src,
                                    bf16* __restrict__ dst, int n) {
  int i = blockIdx.x * blockDim.x + threadIdx.x;
  if (i < n) dst[i] = (bf16)src[i];
}

// ---------------------------------------------------------------------------
// bf16 GEMM: C(MxN f32) = A(MxK bf16, row-major) * B(KxN bf16, row-major)
// 128 threads (4 waves); block computes 128x64; wave computes 32x64
// (2 A-frags x 4 B-frags = 8 WMMAs per 32-deep K step).
// ---------------------------------------------------------------------------
__global__ __launch_bounds__(128) void gemm_bf16_kernel(
    const bf16* __restrict__ A, const bf16* __restrict__ B,
    float* __restrict__ C, int M, int N, int K, int lda, int ldb, int ldc) {
  __shared__ bf16 As[128][32];  // [m][k]
  __shared__ bf16 Bt[64][32];   // [n][k] (transposed stage)

  const int tid  = threadIdx.x;
  const int wv   = tid >> 5;
  const int lane = tid & 31;
  const int m0   = blockIdx.y * 128;
  const int n0   = blockIdx.x * 64;

  v8f c[2][4];
#pragma unroll
  for (int i = 0; i < 2; ++i)
#pragma unroll
    for (int nt = 0; nt < 4; ++nt) c[i][nt] = v8f_zero();

  const int arow = tid >> 2, akch = tid & 3;  // A: 4 iters over +32 rows
  const int bkk = tid >> 3, bnch = tid & 7;   // B: 2 iters over +16 k

  for (int k0 = 0; k0 < K; k0 += 32) {
    // stage A 128x32 via async LDS copies
    {
      const bf16* src = &A[(long)(m0 + arow) * lda + k0 + akch * 8];
#pragma unroll
      for (int it = 0; it < 4; ++it)
        cp_lds_b128(src + (long)it * 32 * lda, &As[arow + it * 32][akch * 8]);
    }
    // stage B 32x64 transposed into Bt[n][k]
    {
      const bf16* src = &B[(long)(k0 + bkk) * ldb + n0 + bnch * 8];
#pragma unroll
      for (int it = 0; it < 2; ++it) {
        union { uint4 q; bf16 h[8]; } u;
        u.q = *(const uint4*)(src + (long)it * 16 * ldb);
#pragma unroll
        for (int j = 0; j < 8; ++j) Bt[bnch * 8 + j][bkk + it * 16] = u.h[j];
      }
    }
    cp_lds_wait();
    __syncthreads();

    const int koffA = (lane >= 16) ? 8 : 0;
    const bf16* ap0 = &As[wv * 32 + (lane & 15)][koffA];
    const bf16* ap1 = &As[wv * 32 + 16 + (lane & 15)][koffA];
    v16bf a0 = ld_frag_2x8(ap0, ap0 + 16);
    v16bf a1 = ld_frag_2x8(ap1, ap1 + 16);
    const int koffB = (lane < 16) ? 0 : 8;
#pragma unroll
    for (int nt = 0; nt < 4; ++nt) {
      v16bf bfg = ld_frag_2x8(&Bt[nt * 16 + (lane & 15)][koffB], &Bt[nt * 16 + (lane & 15)][koffB + 16]);
      c[0][nt] = wmma_bf16(a0, bfg, c[0][nt]);
      c[1][nt] = wmma_bf16(a1, bfg, c[1][nt]);
    }
    __syncthreads();
  }

  const int mAdd = (lane >= 16) ? 8 : 0;
#pragma unroll
  for (int i = 0; i < 2; ++i)
#pragma unroll
    for (int nt = 0; nt < 4; ++nt)
#pragma unroll
      for (int r = 0; r < 8; ++r)
        C[(long)(m0 + wv * 32 + i * 16 + r + mAdd) * ldc + n0 + nt * 16 +
          (lane & 15)] = c[i][nt][r];
}

// ---------------------------------------------------------------------------
// RoPE (half-truncated) + pack q/k/v to bf16, layout [pos][head][128]
// ---------------------------------------------------------------------------
__device__ __forceinline__ void rope_cs(int pos, int j, float& c, float& sn) {
  if (j >= 32) { c = 1.0f; sn = 0.0f; return; }
  float t = (float)j * (1.0f / 31.0f);
  float freq = exp2f(-10.0f * t);  // (1/1024)^t
  float th = (float)pos * freq;
  sincosf(th, &sn, &c);
}

__global__ void rope_pack_kernel(const float* __restrict__ proj,
                                 bf16* __restrict__ QB, bf16* __restrict__ KB,
                                 bf16* __restrict__ VB) {
  int idx = blockIdx.x * blockDim.x + threadIdx.x;
  if (idx >= L_SEQ * 1536) return;
  int pos = idx / 1536;
  int u = idx - pos * 1536;
  const float* row = proj + (long)pos * QKV_DIM;
  if (u < 1024) {
    int hq = u >> 6, j = u & 63;
    float x1 = row[hq * 128 + j], x2 = row[hq * 128 + j + 64];
    float c, sn;
    rope_cs(pos, j, c, sn);
    bf16* q = QB + ((long)pos * N_HQ + hq) * HD;
    q[j]      = (bf16)(x1 * c + x2 * sn);
    q[j + 64] = (bf16)(x2 * c - x1 * sn);
  } else if (u < 1280) {
    int hk = (u - 1024) >> 6, j = u & 63;
    float x1 = row[2048 + hk * 128 + j], x2 = row[2048 + hk * 128 + j + 64];
    float c, sn;
    rope_cs(pos, j, c, sn);
    bf16* k = KB + ((long)pos * N_HK + hk) * HD;
    k[j]      = (bf16)(x1 * c + x2 * sn);
    k[j + 64] = (bf16)(x2 * c - x1 * sn);
  } else {
    int hv = (u - 1280) >> 6, j = u & 63;
    bf16* v = VB + ((long)pos * N_HK + hv) * HD;
    v[j]      = (bf16)row[2560 + hv * 128 + j];
    v[j + 64] = (bf16)row[2560 + hv * 128 + j + 64];
  }
}

// ---------------------------------------------------------------------------
// Flash attention with sink and faithful (inverted) mask:
//   logit = -inf where (q >= k && q < k+WINDOW)
// ---------------------------------------------------------------------------
__global__ __launch_bounds__(128) void attn_kernel(
    const bf16* __restrict__ QB, const bf16* __restrict__ KB,
    const bf16* __restrict__ VB, const float* __restrict__ S,
    bf16* __restrict__ AO) {
  __shared__ bf16 Ks[32][128];    // [key][dim]
  __shared__ bf16 Vt[128][32];    // [dim][key]
  __shared__ bf16 Ps[4][16][32];  // per-wave P scratch

  const int tid  = threadIdx.x;
  const int wv   = tid >> 5;
  const int lane = tid & 31;
  const int hq   = blockIdx.x;
  const int hk   = hq >> 2;
  const int Q0   = blockIdx.y * 64;
  const int q0   = Q0 + wv * 16;
  const float scale = 0.08838834764831845f;  // 1/sqrt(128)

  v16bf qa[4];
  {
    int pos = q0 + (lane & 15);
    int koff = (lane >= 16) ? 8 : 0;
    const bf16* qb = QB + ((long)pos * N_HQ + hq) * HD;
#pragma unroll
    for (int kc = 0; kc < 4; ++kc)
      qa[kc] = ld_frag_2x8(qb + kc * 32 + koff, qb + kc * 32 + koff + 16);
  }

  const float sh = S[hq];
  float rm[8], rl[8];
  v8f acc[8];
#pragma unroll
  for (int r = 0; r < 8; ++r) { rm[r] = sh; rl[r] = 1.0f; }  // sink column
#pragma unroll
  for (int t = 0; t < 8; ++t) acc[t] = v8f_zero();

  const int skey = tid >> 4, sdch = tid & 15;  // 4 iters over +8 keys
  const bf16* kg0 = KB + ((long)skey * N_HK + hk) * HD + sdch * 8;
  const bf16* vg0 = VB + ((long)skey * N_HK + hk) * HD + sdch * 8;

  for (int k0 = 0; k0 < L_SEQ; k0 += 32) {
    if (Q0 >= k0 + 31 && Q0 <= k0 + 960) continue;  // block fully masked

    // stage K tile (async to LDS)
#pragma unroll
    for (int it = 0; it < 4; ++it)
      cp_lds_b128(kg0 + (long)(k0 + it * 8) * (N_HK * HD),
                  &Ks[skey + it * 8][sdch * 8]);
    // stage V tile transposed
#pragma unroll
    for (int it = 0; it < 4; ++it) {
      union { uint4 q; bf16 h[8]; } u;
      u.q = *(const uint4*)(vg0 + (long)(k0 + it * 8) * (N_HK * HD));
#pragma unroll
      for (int j = 0; j < 8; ++j) Vt[sdch * 8 + j][skey + it * 8] = u.h[j];
    }
    cp_lds_wait();
    __syncthreads();

    const bool waveSkip = (q0 >= k0 + 31) && (q0 <= k0 + 1008);
    if (!waveSkip) {
      v8f s0 = v8f_zero(), s1 = v8f_zero();
      const int koffK = (lane < 16) ? 0 : 8;
#pragma unroll
      for (int kc = 0; kc < 4; ++kc) {
        v16bf b0 = ld_frag_2x8(&Ks[lane & 15][kc * 32 + koffK], &Ks[lane & 15][kc * 32 + koffK + 16]);
        v16bf b1 = ld_frag_2x8(&Ks[16 + (lane & 15)][kc * 32 + koffK], &Ks[16 + (lane & 15)][kc * 32 + koffK + 16]);
        s0 = wmma_bf16(qa[kc], b0, s0);
        s1 = wmma_bf16(qa[kc], b1, s1);
      }

      const int n = lane & 15;
      const int mAdd = (lane >= 16) ? 8 : 0;
#pragma unroll
      for (int r = 0; r < 8; ++r) {
        int qp = q0 + r + mAdd;
        int kk0 = k0 + n, kk1 = kk0 + 16;
        float v0 = s0[r] * scale;
        float v1 = s1[r] * scale;
        if (qp >= kk0 && qp < kk0 + WINDOW) v0 = -__builtin_inff();
        if (qp >= kk1 && qp < kk1 + WINDOW) v1 = -__builtin_inff();
        float mx = fmaxf(v0, v1);
#pragma unroll
        for (int off = 1; off < 16; off <<= 1)
          mx = fmaxf(mx, __shfl_xor(mx, off, 32));
        float mnew  = fmaxf(rm[r], mx);
        float alpha = __expf(rm[r] - mnew);
        float p0 = __expf(v0 - mnew);
        float p1 = __expf(v1 - mnew);
        float rs = p0 + p1;
#pragma unroll
        for (int off = 1; off < 16; off <<= 1) rs += __shfl_xor(rs, off, 32);
        rl[r] = rl[r] * alpha + rs;
        rm[r] = mnew;
        Ps[wv][r + mAdd][n]      = (bf16)p0;
        Ps[wv][r + mAdd][16 + n] = (bf16)p1;
#pragma unroll
        for (int t = 0; t < 8; ++t) acc[t][r] *= alpha;
      }
      asm volatile("s_wait_dscnt 0" ::: "memory");

      const bf16* pp = &Ps[wv][lane & 15][(lane >= 16) ? 8 : 0];
      v16bf pa = ld_frag_2x8(pp, pp + 16);
      const int koffV = (lane < 16) ? 0 : 8;
#pragma unroll
      for (int t = 0; t < 8; ++t) {
        v16bf bv = ld_frag_2x8(&Vt[t * 16 + (lane & 15)][koffV], &Vt[t * 16 + (lane & 15)][koffV + 16]);
        acc[t] = wmma_bf16(pa, bv, acc[t]);
      }
    }
    __syncthreads();
  }

  const int mAdd = (lane >= 16) ? 8 : 0;
#pragma unroll
  for (int r = 0; r < 8; ++r) {
    float inv = 1.0f / rl[r];
    long pos = q0 + r + mAdd;
#pragma unroll
    for (int t = 0; t < 8; ++t)
      AO[pos * (N_HQ * HD) + hq * HD + t * 16 + (lane & 15)] =
          (bf16)(acc[t][r] * inv);
  }
}

// ---------------------------------------------------------------------------
// Launch
// ---------------------------------------------------------------------------
extern "C" void kernel_launch(void* const* d_in, const int* in_sizes, int n_in,
                              void* d_out, int out_size, void* d_ws,
                              size_t ws_size, hipStream_t stream) {
  (void)in_sizes; (void)n_in; (void)out_size; (void)ws_size;
  const float* x    = (const float*)d_in[0];
  const float* Wqkv = (const float*)d_in[1];
  const float* Wo   = (const float*)d_in[2];
  const float* s    = (const float*)d_in[3];
  float* out = (float*)d_out;
  char* ws = (char*)d_ws;

  size_t off = 0;
  bf16* xb    = (bf16*)(ws + off); off += (size_t)L_SEQ * D_MODEL * 2;
  bf16* wqkvb = (bf16*)(ws + off); off += (size_t)D_MODEL * QKV_DIM * 2;
  bf16* wob   = (bf16*)(ws + off); off += (size_t)D_MODEL * D_MODEL * 2;
  float* proj = (float*)(ws + off); off += (size_t)L_SEQ * QKV_DIM * 4;
  bf16* qb    = (bf16*)(ws + off); off += (size_t)L_SEQ * N_HQ * HD * 2;
  bf16* kb    = (bf16*)(ws + off); off += (size_t)L_SEQ * N_HK * HD * 2;
  bf16* vb    = (bf16*)(ws + off); off += (size_t)L_SEQ * N_HK * HD * 2;
  bf16* ao    = (bf16*)(ws + off); off += (size_t)L_SEQ * N_HQ * HD * 2;

  const int nx = L_SEQ * D_MODEL;
  const int nw = D_MODEL * QKV_DIM;
  cvt_f32_bf16_kernel<<<(nx + 255) / 256, 256, 0, stream>>>(x, xb, nx);
  cvt_f32_bf16_kernel<<<(nw + 255) / 256, 256, 0, stream>>>(Wqkv, wqkvb, nw);
  cvt_f32_bf16_kernel<<<(nx + 255) / 256, 256, 0, stream>>>(Wo, wob, nx);

  gemm_bf16_kernel<<<dim3(QKV_DIM / 64, L_SEQ / 128), 128, 0, stream>>>(
      xb, wqkvb, proj, L_SEQ, QKV_DIM, D_MODEL, D_MODEL, QKV_DIM, QKV_DIM);

  const int nr = L_SEQ * 1536;
  rope_pack_kernel<<<(nr + 255) / 256, 256, 0, stream>>>(proj, qb, kb, vb);

  attn_kernel<<<dim3(N_HQ, L_SEQ / 64), 128, 0, stream>>>(qb, kb, vb, s, ao);

  gemm_bf16_kernel<<<dim3(D_MODEL / 64, L_SEQ / 128), 128, 0, stream>>>(
      ao, wob, out, L_SEQ, D_MODEL, D_MODEL, D_MODEL, D_MODEL, D_MODEL);
}
